// BiLSTM_CRF_24653112279157
// MI455X (gfx1250) — matrix operand order repaired and checked by reading, not compiled, not run on hardware
//
#include <hip/hip_runtime.h>
#include <hip/hip_bf16.h>
#include <math.h>

#define B_ 64
#define S_ 256
#define D_ 256
#define H_ 512
#define T_ 48
#define EPS_ 1e-5f

typedef __bf16 bf16;
typedef __attribute__((ext_vector_type(8)))  __bf16 v8bf;
typedef __attribute__((ext_vector_type(16))) __bf16 v16bf;
typedef __attribute__((ext_vector_type(8)))  float  v8f;
typedef __attribute__((ext_vector_type(4)))  unsigned int u32x4;
typedef __attribute__((ext_vector_type(8)))  int i32x8;
typedef __attribute__((ext_vector_type(4)))  int i32x4;

#if defined(__has_builtin)
#if __has_builtin(__builtin_amdgcn_tensor_load_to_lds) && __has_builtin(__builtin_amdgcn_s_wait_tensorcnt)
#define USE_TDM 1
#endif
#endif
#ifndef USE_TDM
#define USE_TDM 0
#endif

__device__ __forceinline__ v16bf cat16(v8bf lo, v8bf hi) {
  return __builtin_shufflevector(lo, hi, 0,1,2,3,4,5,6,7,8,9,10,11,12,13,14,15);
}
__device__ __forceinline__ v8f wmma_bf16(v16bf a, v16bf b, v8f c) {
  // 8 args: (neg_a, A, neg_b, B, c_mod, C, reuse_a, reuse_b)
  return __builtin_amdgcn_wmma_f32_16x16x32_bf16(false, a, false, b, (short)0, c, false, false);
}
__device__ __forceinline__ float sigm(float x) { return 1.0f / (1.0f + __expf(-x)); }

#if USE_TDM
// TDM: load a rows x 256-element (bf16) 2D tile into LDS at lds_off, with
// HW padding of 4 DWORDs (16B) after every 128 DWORDs (512B = one 256-elem row),
// matching the As[64][264] padded layout.
__device__ __forceinline__ void tdm_load_2d(unsigned lds_off, const bf16* gsrc,
                                            unsigned rows, unsigned row_stride_elems) {
  unsigned long long ga = (unsigned long long)(size_t)gsrc;
  const unsigned td0 = 0x40000000u, td1 = 0x40000000u;   // huge tensor dims: no OOB clamp
  u32x4 g0;
  g0[0] = 1u;                                            // count=1, user descriptor
  g0[1] = lds_off;                                       // LDS byte address
  g0[2] = (unsigned)(ga & 0xFFFFFFFFu);                  // global_addr[31:0]
  g0[3] = (unsigned)((ga >> 32) & 0x1FFFFFFu) | (2u << 30); // global_addr[56:32] | type=2
  i32x8 g1;
  g1[0] = (int)((1u << 16)      // data_size = 2 bytes
              | (1u << 20)      // pad_enable
              | (6u << 22)      // pad_interval: 128 DWORDs
              | (3u << 25));    // pad_amount: 4 DWORDs
  g1[1] = (int)((td0 & 0xFFFFu) << 16);                              // tensor_dim0[15:0]
  g1[2] = (int)(((td0 >> 16) & 0xFFFFu) | ((td1 & 0xFFFFu) << 16));  // td0[31:16] | td1[15:0]
  g1[3] = (int)(((td1 >> 16) & 0xFFFFu) | (256u << 16));             // td1[31:16] | tile_dim0=256
  g1[4] = (int)(rows & 0xFFFFu);                                     // tile_dim1, tile_dim2=0
  g1[5] = (int)row_stride_elems;                                     // tensor_dim0_stride[31:0]
  g1[6] = 0;                                                         // stride0[47:32] | stride1[15:0]
  g1[7] = 0;
  i32x4 z4 = {0, 0, 0, 0};                                           // 2D tensor: groups 2/3 unused
  i32x8 z8 = {0, 0, 0, 0, 0, 0, 0, 0};
  __builtin_amdgcn_tensor_load_to_lds(g0, g1, z4, z4, z8, 0);
}
#endif

// ---------------- conversion / packing ----------------

__global__ void k_f32_to_bf16(const float* __restrict__ src, bf16* __restrict__ dst, int n) {
  int i = blockIdx.x * 256 + threadIdx.x;
  if (i < n) dst[i] = (bf16)src[i];
}

__global__ void k_bias_sum(const float* __restrict__ a, const float* __restrict__ b,
                           float* __restrict__ o, int n) {
  int i = blockIdx.x * 256 + threadIdx.x;
  if (i < n) o[i] = a[i] + b[i];
}

// Pack [Wih | Whh] (4H x (Din+H)) into WMMA B-fragment order with gate-permuted N.
// Linear index: ((kt*128 + pt)*32 + lane)*16 + e
//   k = kt*32 + (lane>>4)*8 + (e&7) + (e>>3)*16
//   pt: blk(3b) gate(2b) jb(2b); n = gate*H + blk*64 + jb*16 + (lane&15)
__global__ void k_pack_lstm(const float* __restrict__ Wih, const float* __restrict__ Whh,
                            bf16* __restrict__ dst, int Din, int total) {
  int idx = blockIdx.x * 256 + threadIdx.x;
  if (idx >= total) return;
  int e = idx & 15, lane = (idx >> 4) & 31, t = idx >> 9;
  int pt = t & 127, kt = t >> 7;
  int k = kt * 32 + ((lane >> 4) << 3) + (e & 7) + ((e >> 3) << 4);
  int blk = pt >> 4, gate = (pt >> 2) & 3, jb = pt & 3;
  int n = gate * H_ + blk * 64 + jb * 16 + (lane & 15);
  float v = (k < Din) ? Wih[(size_t)n * Din + k] : Whh[(size_t)n * H_ + (k - Din)];
  dst[idx] = (bf16)v;
}

// Pack fc W (N x K) as B = W^T in fragment order (no N permutation).
__global__ void k_pack_fc(const float* __restrict__ W, bf16* __restrict__ dst,
                          int N, int K, const float* __restrict__ gamma,
                          const float* __restrict__ var, int total) {
  int idx = blockIdx.x * 256 + threadIdx.x;
  if (idx >= total) return;
  int NPT = N >> 4;
  int e = idx & 15, lane = (idx >> 4) & 31, t = idx >> 9;
  int pt = t % NPT, kt = t / NPT;
  int k = kt * 32 + ((lane >> 4) << 3) + (e & 7) + ((e >> 3) << 4);
  int n = pt * 16 + (lane & 15);
  float v = W[(size_t)n * K + k];
  if (gamma) v *= gamma[k] * rsqrtf(var[k] + EPS_);
  dst[idx] = (bf16)v;
}

__global__ void k_fold_fc1_bias(const float* __restrict__ fc1w, const float* __restrict__ fc1b,
                                const float* __restrict__ gamma, const float* __restrict__ beta,
                                const float* __restrict__ mean, const float* __restrict__ var,
                                float* __restrict__ fb) {
  int n = blockIdx.x * 256 + threadIdx.x;
  if (n >= H_) return;
  float s = fc1b[n];
  for (int k = 0; k < 2 * H_; ++k) {
    float sc = gamma[k] * rsqrtf(var[k] + EPS_);
    float sh = beta[k] - mean[k] * sc;
    s += fc1w[(size_t)n * (2 * H_) + k] * sh;
  }
  fb[n] = s;
}

__global__ void k_init_state(bf16* __restrict__ h, float* __restrict__ c, int nh, int nc) {
  int i = blockIdx.x * 256 + threadIdx.x;
  if (i < nh) h[i] = (bf16)0.0f;
  if (i < nc) c[i] = 0.0f;
}

// ---------------- LSTM step (both directions per launch) ----------------
// grid: 16 blocks = 2 dirs x 8 column-blocks (64 hidden cols each), 256 threads (8 waves).
// Wave w: M-tile (w&3), jb-pair (w>>2). Each wave owns i/f/g/o tiles for its columns.
__global__ void __launch_bounds__(256) k_lstm_step(
    const bf16* __restrict__ x, int Din,
    const bf16* __restrict__ Bp_f, const bf16* __restrict__ Bp_b,
    const float* __restrict__ bs_f, const float* __restrict__ bs_b,
    bf16* __restrict__ hbase, float* __restrict__ cbase,
    bf16* __restrict__ out, int t)
{
  __shared__ bf16 As[64][264];   // 64 rows x 256-K chunk, +16B row pad (TDM pad feature)
  const int dir = blockIdx.x >> 3;
  const int blk = blockIdx.x & 7;
  const int tid = threadIdx.x;
  const int lane = tid & 31, wave = tid >> 5;
  const int mt = wave & 3, jbp = wave >> 2;
  const int half = lane >> 4, ln = lane & 15;

  const int KT = Din + H_;
  const int nchunks = KT >> 8;
  const int time = dir ? (S_ - 1 - t) : t;
  const bf16* Bp = dir ? Bp_b : Bp_f;
  const float* bs = dir ? bs_b : bs_f;
  const bf16* hread  = hbase + (size_t)(dir * 2 + (t & 1)) * (B_ * H_);
  bf16*       hwrite = hbase + (size_t)(dir * 2 + ((t + 1) & 1)) * (B_ * H_);
  float*      cell   = cbase + (size_t)dir * (B_ * H_);

#if USE_TDM
  // low 32 bits of a generic LDS pointer = byte offset within LDS
  const unsigned lds_base = (unsigned)(size_t)&As[0][0];
#endif

  v8f acc[2][4] = {};

  for (int ch = 0; ch < nchunks; ++ch) {
    const int kbase = ch << 8;
    __syncthreads();
#if USE_TDM
    if (wave == 0) {
      const bf16* src;
      unsigned stride;
      if (kbase < Din) { src = x + (size_t)time * Din + kbase; stride = (unsigned)(S_ * Din); }
      else             { src = hread + (kbase - Din);          stride = (unsigned)H_; }
      tdm_load_2d(lds_base, src, 64u, stride);
      __builtin_amdgcn_s_wait_tensorcnt(0);
    }
#else
    for (int i = tid; i < 64 * 32; i += 256) {
      int r = i >> 5, cb = i & 31;
      const bf16* src;
      if (kbase < Din)
        src = x + ((size_t)r * S_ + time) * Din + kbase + cb * 8;
      else
        src = hread + (size_t)r * H_ + (kbase - Din) + cb * 8;
      *(v8bf*)&As[r][cb * 8] = *(const v8bf*)src;
    }
#endif
    __syncthreads();
    for (int kl = 0; kl < 8; ++kl) {
      const int ktg = (ch << 3) + kl;
      const int bc = kl * 32 + half * 8;
      v8bf alo = *(const v8bf*)&As[mt * 16 + ln][bc];
      v8bf ahi = *(const v8bf*)&As[mt * 16 + ln][bc + 16];
      v16bf afrag = cat16(alo, ahi);
      #pragma unroll
      for (int jbl = 0; jbl < 2; ++jbl) {
        const int jb = jbp * 2 + jbl;
        #pragma unroll
        for (int g = 0; g < 4; ++g) {
          const int pt = blk * 16 + g * 4 + jb;
          const bf16* bp = Bp + ((((size_t)ktg * 128 + pt) * 32 + lane) << 4);
          v8bf blo = *(const v8bf*)bp;
          v8bf bhi = *(const v8bf*)(bp + 8);
          acc[jbl][g] = wmma_bf16(afrag, cat16(blo, bhi), acc[jbl][g]);
        }
      }
    }
  }

  // in-register gate nonlinearity + state update
  #pragma unroll
  for (int jbl = 0; jbl < 2; ++jbl) {
    const int jb = jbp * 2 + jbl;
    const int j = blk * 64 + jb * 16 + ln;
    #pragma unroll
    for (int vr = 0; vr < 8; ++vr) {
      const int b = mt * 16 + half * 8 + vr;
      float gi = acc[jbl][0][vr] + bs[j];
      float gf = acc[jbl][1][vr] + bs[H_ + j];
      float gg = acc[jbl][2][vr] + bs[2 * H_ + j];
      float go = acc[jbl][3][vr] + bs[3 * H_ + j];
      float cp = cell[(size_t)b * H_ + j];
      float cn = sigm(gf) * cp + sigm(gi) * tanhf(gg);
      cell[(size_t)b * H_ + j] = cn;
      float hv = sigm(go) * tanhf(cn);
      bf16 hb = (bf16)hv;
      hwrite[(size_t)b * H_ + j] = hb;
      out[((size_t)b * S_ + time) * (2 * H_) + (size_t)dir * H_ + j] = hb;
    }
  }
}

// ---------------- generic WMMA GEMM (head) ----------------
__global__ void __launch_bounds__(256) k_gemm(
    const bf16* __restrict__ A, const bf16* __restrict__ Bp,
    const float* __restrict__ bias, void* __restrict__ Cout,
    int M, int N, int K, int relu, int out_bf16)
{
  const int lane = threadIdx.x & 31;
  const int gw = blockIdx.x * 8 + (threadIdx.x >> 5);
  const int Ntiles = N >> 4;
  const int total = (M >> 4) * Ntiles;
  if (gw >= total) return;
  const int mt = gw / Ntiles, nt = gw % Ntiles;
  const int half = lane >> 4, ln = lane & 15;
  const int KT = K >> 5;
  v8f acc = {};
  const bf16* arow = A + (size_t)(mt * 16 + ln) * K + half * 8;
  for (int kt = 0; kt < KT; ++kt) {
    v8bf alo = *(const v8bf*)(arow + kt * 32);
    v8bf ahi = *(const v8bf*)(arow + kt * 32 + 16);
    const bf16* bp = Bp + ((((size_t)kt * Ntiles + nt) * 32 + lane) << 4);
    v8bf blo = *(const v8bf*)bp;
    v8bf bhi = *(const v8bf*)(bp + 8);
    acc = wmma_bf16(cat16(alo, ahi), cat16(blo, bhi), acc);
  }
  const int n = nt * 16 + ln;
  #pragma unroll
  for (int vr = 0; vr < 8; ++vr) {
    const int m = mt * 16 + half * 8 + vr;
    float v = acc[vr] + (bias ? bias[n] : 0.0f);
    if (relu) v = fmaxf(v, 0.0f);
    if (out_bf16) ((bf16*)Cout)[(size_t)m * N + n] = (bf16)v;
    else          ((float*)Cout)[(size_t)m * N + n] = v;
  }
}

// ---------------- CRF ----------------
__global__ void k_crf_num(const float* __restrict__ em, const int* __restrict__ tags,
                          const float* __restrict__ start, const float* __restrict__ endv,
                          const float* __restrict__ trans, float* __restrict__ num)
{
  __shared__ float red[256];
  const int b = blockIdx.x, tid = threadIdx.x;
  float s = 0.0f;
  for (int t = tid; t < S_; t += 256) {
    int tg = tags[b * S_ + t];
    s += em[((size_t)b * S_ + t) * T_ + tg];
    if (t >= 1) {
      int tp = tags[b * S_ + t - 1];
      s += trans[tp * T_ + tg];
    }
  }
  red[tid] = s;
  __syncthreads();
  for (int o = 128; o > 0; o >>= 1) {
    if (tid < o) red[tid] += red[tid + o];
    __syncthreads();
  }
  if (tid == 0)
    num[b] = red[0] + start[tags[b * S_]] + endv[tags[b * S_ + S_ - 1]];
}

__global__ void k_crf_logz(const float* __restrict__ em, const float* __restrict__ start,
                           const float* __restrict__ endv, const float* __restrict__ trans,
                           float* __restrict__ logz)
{
  __shared__ float ltr[T_ * T_];
  __shared__ float alpha[T_];
  const int b = blockIdx.x, tid = threadIdx.x;   // 64 threads
  for (int i = tid; i < T_ * T_; i += 64) ltr[i] = trans[i];
  if (tid < T_) alpha[tid] = start[tid] + em[((size_t)b * S_) * T_ + tid];
  __syncthreads();
  for (int t = 1; t < S_; ++t) {
    float na = 0.0f;
    if (tid < T_) {
      float m = -1e30f;
      for (int i = 0; i < T_; ++i) m = fmaxf(m, alpha[i] + ltr[i * T_ + tid]);
      float s = 0.0f;
      for (int i = 0; i < T_; ++i) s += __expf(alpha[i] + ltr[i * T_ + tid] - m);
      na = m + __logf(s) + em[((size_t)b * S_ + t) * T_ + tid];
    }
    __syncthreads();
    if (tid < T_) alpha[tid] = na;
    __syncthreads();
  }
  if (tid == 0) {
    float m = -1e30f;
    for (int i = 0; i < T_; ++i) m = fmaxf(m, alpha[i] + endv[i]);
    float s = 0.0f;
    for (int i = 0; i < T_; ++i) s += __expf(alpha[i] + endv[i] - m);
    logz[b] = m + __logf(s);
  }
}

__global__ void k_final(const float* __restrict__ num, const float* __restrict__ logz,
                        float* __restrict__ out) {
  __shared__ float red[64];
  int tid = threadIdx.x;
  red[tid] = num[tid] - logz[tid];
  __syncthreads();
  for (int o = 32; o > 0; o >>= 1) {
    if (tid < o) red[tid] += red[tid + o];
    __syncthreads();
  }
  if (tid == 0) out[0] = -red[0] / (float)B_;
}

// ---------------- host ----------------
extern "C" void kernel_launch(void* const* d_in, const int* in_sizes, int n_in,
                              void* d_out, int out_size, void* d_ws, size_t ws_size,
                              hipStream_t stream)
{
  (void)in_sizes; (void)n_in; (void)out_size; (void)ws_size;

  const float* inputs  = (const float*)d_in[0];
  const int*   tags    = (const int*)d_in[1];
  const float* Wih_l0f = (const float*)d_in[2];
  const float* Whh_l0f = (const float*)d_in[3];
  const float* bih_l0f = (const float*)d_in[4];
  const float* bhh_l0f = (const float*)d_in[5];
  const float* Wih_l0b = (const float*)d_in[6];
  const float* Whh_l0b = (const float*)d_in[7];
  const float* bih_l0b = (const float*)d_in[8];
  const float* bhh_l0b = (const float*)d_in[9];
  const float* Wih_l1f = (const float*)d_in[10];
  const float* Whh_l1f = (const float*)d_in[11];
  const float* bih_l1f = (const float*)d_in[12];
  const float* bhh_l1f = (const float*)d_in[13];
  const float* Wih_l1b = (const float*)d_in[14];
  const float* Whh_l1b = (const float*)d_in[15];
  const float* bih_l1b = (const float*)d_in[16];
  const float* bhh_l1b = (const float*)d_in[17];
  const float* bn_gamma = (const float*)d_in[18];
  const float* bn_beta  = (const float*)d_in[19];
  const float* bn_mean  = (const float*)d_in[20];
  const float* bn_var   = (const float*)d_in[21];
  const float* fc1_w = (const float*)d_in[22];
  const float* fc1_b = (const float*)d_in[23];
  const float* fc2_w = (const float*)d_in[24];
  const float* fc2_b = (const float*)d_in[25];
  const float* fc3_w = (const float*)d_in[26];
  const float* fc3_b = (const float*)d_in[27];
  const float* crf_start = (const float*)d_in[28];
  const float* crf_end   = (const float*)d_in[29];
  const float* crf_trans = (const float*)d_in[30];

  char* ws = (char*)d_ws;
  size_t off = 0;
  auto alloc = [&](size_t bytes) -> void* {
    void* p = ws + off;
    off += (bytes + 255) & ~(size_t)255;
    return p;
  };

  bf16*  xbf   = (bf16*) alloc((size_t)B_ * S_ * D_ * 2);
  bf16*  out0  = (bf16*) alloc((size_t)B_ * S_ * 2 * H_ * 2);
  bf16*  out1  = (bf16*) alloc((size_t)B_ * S_ * 2 * H_ * 2);
  bf16*  fc1o  = (bf16*) alloc((size_t)B_ * S_ * H_ * 2);
  bf16*  fc2o  = (bf16*) alloc((size_t)B_ * S_ * (H_ / 2) * 2);
  float* em    = (float*)alloc((size_t)B_ * S_ * T_ * 4);
  bf16*  pk_l0 = (bf16*) alloc((size_t)2 * 4 * H_ * (D_ + H_) * 2);
  bf16*  pk_l1 = (bf16*) alloc((size_t)2 * 4 * H_ * (3 * H_) * 2);
  bf16*  pk_f1 = (bf16*) alloc((size_t)H_ * 2 * H_ * 2);
  bf16*  pk_f2 = (bf16*) alloc((size_t)(H_ / 2) * H_ * 2);
  bf16*  pk_f3 = (bf16*) alloc((size_t)T_ * (H_ / 2) * 2);
  float* fb1   = (float*)alloc(H_ * 4);
  float* bs_l0f = (float*)alloc(4 * H_ * 4);
  float* bs_l0b = (float*)alloc(4 * H_ * 4);
  float* bs_l1f = (float*)alloc(4 * H_ * 4);
  float* bs_l1b = (float*)alloc(4 * H_ * 4);
  bf16*  hbuf  = (bf16*) alloc((size_t)4 * B_ * H_ * 2);   // 2 dirs x 2 (double-buffer)
  float* cbuf  = (float*)alloc((size_t)2 * B_ * H_ * 4);
  float* numb  = (float*)alloc(B_ * 4);
  float* lgz   = (float*)alloc(B_ * 4);

  // input -> bf16
  {
    int n = B_ * S_ * D_;
    k_f32_to_bf16<<<(n + 255) / 256, 256, 0, stream>>>(inputs, xbf, n);
  }
  // weight packing + bias folding
  {
    int tot0 = 4 * H_ * (D_ + H_);
    k_pack_lstm<<<tot0 / 256, 256, 0, stream>>>(Wih_l0f, Whh_l0f, pk_l0, D_, tot0);
    k_pack_lstm<<<tot0 / 256, 256, 0, stream>>>(Wih_l0b, Whh_l0b, pk_l0 + tot0, D_, tot0);
    int tot1 = 4 * H_ * (3 * H_);
    k_pack_lstm<<<tot1 / 256, 256, 0, stream>>>(Wih_l1f, Whh_l1f, pk_l1, 2 * H_, tot1);
    k_pack_lstm<<<tot1 / 256, 256, 0, stream>>>(Wih_l1b, Whh_l1b, pk_l1 + tot1, 2 * H_, tot1);
    k_pack_fc<<<(H_ * 2 * H_) / 256, 256, 0, stream>>>(fc1_w, pk_f1, H_, 2 * H_, bn_gamma, bn_var, H_ * 2 * H_);
    k_pack_fc<<<((H_ / 2) * H_) / 256, 256, 0, stream>>>(fc2_w, pk_f2, H_ / 2, H_, nullptr, nullptr, (H_ / 2) * H_);
    k_pack_fc<<<(T_ * (H_ / 2)) / 256, 256, 0, stream>>>(fc3_w, pk_f3, T_, H_ / 2, nullptr, nullptr, T_ * (H_ / 2));
    k_fold_fc1_bias<<<2, 256, 0, stream>>>(fc1_w, fc1_b, bn_gamma, bn_beta, bn_mean, bn_var, fb1);
    k_bias_sum<<<8, 256, 0, stream>>>(bih_l0f, bhh_l0f, bs_l0f, 4 * H_);
    k_bias_sum<<<8, 256, 0, stream>>>(bih_l0b, bhh_l0b, bs_l0b, 4 * H_);
    k_bias_sum<<<8, 256, 0, stream>>>(bih_l1f, bhh_l1f, bs_l1f, 4 * H_);
    k_bias_sum<<<8, 256, 0, stream>>>(bih_l1b, bhh_l1b, bs_l1b, 4 * H_);
  }
  // layer 0 BiLSTM
  k_init_state<<<512, 256, 0, stream>>>(hbuf, cbuf, 4 * B_ * H_, 2 * B_ * H_);
  for (int t = 0; t < S_; ++t)
    k_lstm_step<<<16, 256, 0, stream>>>(xbf, D_, pk_l0, pk_l0 + 4 * H_ * (D_ + H_),
                                        bs_l0f, bs_l0b, hbuf, cbuf, out0, t);
  // layer 1 BiLSTM
  k_init_state<<<512, 256, 0, stream>>>(hbuf, cbuf, 4 * B_ * H_, 2 * B_ * H_);
  for (int t = 0; t < S_; ++t)
    k_lstm_step<<<16, 256, 0, stream>>>(out0, 2 * H_, pk_l1, pk_l1 + 4 * H_ * (3 * H_),
                                        bs_l1f, bs_l1b, hbuf, cbuf, out1, t);
  // head: BN folded into fc1
  const int Mrows = B_ * S_;
  k_gemm<<<(Mrows / 16) * (H_ / 16) / 8, 256, 0, stream>>>(out1, pk_f1, fb1, fc1o, Mrows, H_, 2 * H_, 1, 1);
  k_gemm<<<(Mrows / 16) * ((H_ / 2) / 16) / 8, 256, 0, stream>>>(fc1o, pk_f2, fc2_b, fc2o, Mrows, H_ / 2, H_, 1, 1);
  k_gemm<<<(Mrows / 16) * (T_ / 16) / 8, 256, 0, stream>>>(fc2o, pk_f3, fc3_b, em, Mrows, T_, H_ / 2, 0, 0);
  // CRF
  k_crf_num<<<B_, 256, 0, stream>>>(em, tags, crf_start, crf_end, crf_trans, numb);
  k_crf_logz<<<B_, 64, 0, stream>>>(em, crf_start, crf_end, crf_trans, lgz);
  k_final<<<1, 64, 0, stream>>>(numb, lgz, (float*)d_out);
}
